// MoE_27530740368053
// MI455X (gfx1250) — matrix operand order repaired and checked by reading, not compiled, not run on hardware
//
#include <hip/hip_runtime.h>

// MoE: B=4 L=2048 D=1024 E=8 K=2 DFF=2048, T = 8192 tokens.
// Memory-bound (~270 MB HBM vs a few us of WMMA compute) -> bf16 tensor path
// with f32 accumulate; h intermediate staged bf16 in workspace (L2-resident,
// 33MB). moe_down uses the Tensor Data Mover to DMA h tiles into LDS.

#define D_   1024
#define E_   8
#define DFF_ 2048
#define T_   8192

#ifndef __has_builtin
#define __has_builtin(x) 0
#endif

typedef __attribute__((ext_vector_type(8)))  float  v8f;
typedef __attribute__((ext_vector_type(16))) __bf16 v16bf;

#if __has_builtin(__builtin_amdgcn_tensor_load_to_lds)
#define USE_TDM 1
typedef unsigned int u32x4 __attribute__((ext_vector_type(4)));
typedef int          i32x4 __attribute__((ext_vector_type(4)));
typedef int          i32x8 __attribute__((ext_vector_type(8)));
#else
#define USE_TDM 0
#endif

union BFrag {            // one WMMA bf16 operand: 16 bf16 = 32 bytes
    v16bf v;
    uint4 q[2];
};
union BPack {            // 16 f32 -> 16 bf16 staging helper
    __bf16 b[16];
    uint4  q[2];
};

// ---------------------------------------------------------------- router ----
__global__ __launch_bounds__(256) void moe_router(const float* __restrict__ x,
                                                  const float* __restrict__ Wr,
                                                  float* __restrict__ combine) {
    const int wave = threadIdx.x >> 5;
    const int lane = threadIdx.x & 31;
    const int t = blockIdx.x * 8 + wave;

    float acc[E_];
#pragma unroll
    for (int e = 0; e < E_; ++e) acc[e] = 0.f;

    const float* xr = x + (size_t)t * D_;
    for (int d = lane; d < D_; d += 32) {
        const float xv = xr[d];
        const float* wr = Wr + (size_t)d * E_;
#pragma unroll
        for (int e = 0; e < E_; ++e) acc[e] += xv * wr[e];
    }
#pragma unroll
    for (int off = 16; off > 0; off >>= 1) {
#pragma unroll
        for (int e = 0; e < E_; ++e) acc[e] += __shfl_xor(acc[e], off, 32);
    }
    // softmax over 8 logits (all lanes hold the full reduced vector)
    float m = acc[0];
#pragma unroll
    for (int e = 1; e < E_; ++e) m = fmaxf(m, acc[e]);
    float p[E_], s = 0.f;
#pragma unroll
    for (int e = 0; e < E_; ++e) { p[e] = __expf(acc[e] - m); s += p[e]; }
    const float inv = 1.f / s;
#pragma unroll
    for (int e = 0; e < E_; ++e) p[e] *= inv;
    // top-2 (first index wins ties, like lax.top_k)
    int i1 = 0;
#pragma unroll
    for (int e = 1; e < E_; ++e) if (p[e] > p[i1]) i1 = e;
    int i2 = (i1 == 0) ? 1 : 0;
#pragma unroll
    for (int e = 0; e < E_; ++e) if (e != i1 && p[e] > p[i2]) i2 = e;
    const float kw = p[i1] + p[i2];
    const float w1 = p[i1] / kw, w2 = p[i2] / kw;
    if (lane < E_) {
        const float v = (lane == i1) ? w1 : ((lane == i2) ? w2 : 0.f);
        combine[(size_t)t * E_ + lane] = v;
    }
}

// ------------------------------------------------------------- gate / up ----
// h[t,f] = relu(x@Wg)[t,f] * (x@Wu)[t,f] * combine[t,e], stored bf16.
// Block tile: 128 tokens x 64 f-cols; 8 waves in 4(M) x 2(N) grid, each wave
// owns 2x2 WMMA C-tiles per matrix (gate and up).
__global__ __launch_bounds__(256) void moe_gate_up(const float* __restrict__ x,
                                                   const float* __restrict__ Wg,
                                                   const float* __restrict__ Wu,
                                                   const float* __restrict__ combine,
                                                   __bf16* __restrict__ h,
                                                   int e) {
    __shared__ __bf16 Axb[128][40];   // [token][k]   row = 80B, 16B-aligned
    __shared__ __bf16 Bgt[64][40];    // [n][k] transposed
    __shared__ __bf16 But[64][40];

    const int tid  = threadIdx.x;
    const int lane = tid & 31;
    const int w    = tid >> 5;
    const int wm   = w & 3;           // 0..3 -> M offset *32
    const int wn   = w >> 2;          // 0..1 -> N offset *32
    const int tokBase = blockIdx.y * 128;
    const int fBase   = blockIdx.x * 64;

    const float* Wge = Wg + (size_t)e * D_ * DFF_;
    const float* Wue = Wu + (size_t)e * D_ * DFF_;

    v8f ag[2][2], au[2][2];
    const v8f vzero = {0.f, 0.f, 0.f, 0.f, 0.f, 0.f, 0.f, 0.f};
#pragma unroll
    for (int i = 0; i < 2; ++i)
#pragma unroll
        for (int j = 0; j < 2; ++j) { ag[i][j] = vzero; au[i][j] = vzero; }

    // staging index plans
    const int arow  = tid >> 1;            // 0..127
    const int ahalf = (tid & 1) * 16;      // k-chunk of 16
    const int bk    = tid >> 3;            // 0..31
    const int bn0   = (tid & 7) * 8;       // 8 n-columns

    for (int kBase = 0; kBase < D_; kBase += 32) {
        __syncthreads();
        // ---- stage x tile (f32 -> bf16) : 128 x 32
        {
            const float* src = &x[(size_t)(tokBase + arow) * D_ + kBase + ahalf];
            BPack t;
#pragma unroll
            for (int q = 0; q < 4; ++q) {
                const float4 f = *(const float4*)(src + q * 4);
                t.b[q * 4 + 0] = (__bf16)f.x; t.b[q * 4 + 1] = (__bf16)f.y;
                t.b[q * 4 + 2] = (__bf16)f.z; t.b[q * 4 + 3] = (__bf16)f.w;
            }
            *(uint4*)&Axb[arow][ahalf + 0] = t.q[0];
            *(uint4*)&Axb[arow][ahalf + 8] = t.q[1];
        }
        // ---- stage Wg/Wu tiles (f32 -> bf16, transposed [n][k]) : 32 x 64
        {
            const float* gs = &Wge[(size_t)(kBase + bk) * DFF_ + fBase + bn0];
            const float* us = &Wue[(size_t)(kBase + bk) * DFF_ + fBase + bn0];
            const float4 g0 = *(const float4*)(gs + 0);
            const float4 g1 = *(const float4*)(gs + 4);
            const float4 u0 = *(const float4*)(us + 0);
            const float4 u1 = *(const float4*)(us + 4);
            Bgt[bn0 + 0][bk] = (__bf16)g0.x; Bgt[bn0 + 1][bk] = (__bf16)g0.y;
            Bgt[bn0 + 2][bk] = (__bf16)g0.z; Bgt[bn0 + 3][bk] = (__bf16)g0.w;
            Bgt[bn0 + 4][bk] = (__bf16)g1.x; Bgt[bn0 + 5][bk] = (__bf16)g1.y;
            Bgt[bn0 + 6][bk] = (__bf16)g1.z; Bgt[bn0 + 7][bk] = (__bf16)g1.w;
            But[bn0 + 0][bk] = (__bf16)u0.x; But[bn0 + 1][bk] = (__bf16)u0.y;
            But[bn0 + 2][bk] = (__bf16)u0.z; But[bn0 + 3][bk] = (__bf16)u0.w;
            But[bn0 + 4][bk] = (__bf16)u1.x; But[bn0 + 5][bk] = (__bf16)u1.y;
            But[bn0 + 6][bk] = (__bf16)u1.z; But[bn0 + 7][bk] = (__bf16)u1.w;
            if (kBase + 32 < D_) {   // global_prefetch_b8 on next weight chunk
                __builtin_prefetch(gs + 32 * DFF_, 0, 1);
                __builtin_prefetch(us + 32 * DFF_, 0, 1);
            }
        }
        __syncthreads();

        // ---- WMMA: per wave 2 A-frags, 2 Bg-frags, 2 Bu-frags -> 8 wmma
        const int rsel = lane & 15;
        const int kb   = (lane >> 4) * 8;   // A half-wave K offset (pairs)
        const int kbb  = (lane >> 4) * 8;   // B half-wave K offset
        BFrag a[2], bg[2], bu[2];
#pragma unroll
        for (int i = 0; i < 2; ++i) {
            const int r = wm * 32 + i * 16 + rsel;
            a[i].q[0] = *(const uint4*)&Axb[r][kb];
            a[i].q[1] = *(const uint4*)&Axb[r][16 + kb];
        }
#pragma unroll
        for (int j = 0; j < 2; ++j) {
            const int n = wn * 32 + j * 16 + rsel;
            bg[j].q[0] = *(const uint4*)&Bgt[n][kbb];
            bg[j].q[1] = *(const uint4*)&Bgt[n][kbb + 16];
            bu[j].q[0] = *(const uint4*)&But[n][kbb];
            bu[j].q[1] = *(const uint4*)&But[n][kbb + 16];
        }
#pragma unroll
        for (int i = 0; i < 2; ++i)
#pragma unroll
            for (int j = 0; j < 2; ++j) {
                ag[i][j] = __builtin_amdgcn_wmma_f32_16x16x32_bf16(
                    false, a[i].v, false, bg[j].v, (short)0, ag[i][j], false, false);
                au[i][j] = __builtin_amdgcn_wmma_f32_16x16x32_bf16(
                    false, a[i].v, false, bu[j].v, (short)0, au[i][j], false, false);
            }
    }

    // ---- epilogue: relu(g)*u * combine -> bf16 h
    const int rsel  = lane & 15;
    const int rhalf = (lane >> 4) * 8;
#pragma unroll
    for (int i = 0; i < 2; ++i) {
        float ce[8];
#pragma unroll
        for (int v = 0; v < 8; ++v) {
            const int row = tokBase + wm * 32 + i * 16 + rhalf + v;
            ce[v] = combine[(size_t)row * E_ + e];
        }
#pragma unroll
        for (int j = 0; j < 2; ++j) {
#pragma unroll
            for (int v = 0; v < 8; ++v) {
                const int row = tokBase + wm * 32 + i * 16 + rhalf + v;
                const int col = fBase + wn * 32 + j * 16 + rsel;
                const float g = ag[i][j][v];
                const float u = au[i][j][v];
                h[(size_t)row * DFF_ + col] = (__bf16)(fmaxf(g, 0.f) * u * ce[v]);
            }
        }
    }
}

// ------------------------------------------------------------------ down ----
// out[t,d] (+)= (h @ Wd[e])[t,d].  h already carries the combine weight.
// The bf16 h tile (128x32, 8KB) is a pure 2D strided copy -> Tensor Data
// Mover DMA with LDS row padding; Wd needs f32->bf16 so it stays on VALU.
__global__ __launch_bounds__(256) void moe_down(const __bf16* __restrict__ h,
                                                const float* __restrict__ Wd,
                                                float* __restrict__ out,
                                                int e, int first) {
    __shared__ __bf16 Ah[128][40];    // [token][k], rows padded 64B -> 80B
    __shared__ __bf16 Bt[64][40];     // [n][k] transposed, f32->bf16

    const int tid  = threadIdx.x;
    const int lane = tid & 31;
    const int w    = tid >> 5;
    const int wm   = w & 3;
    const int wn   = w >> 2;
    const int tokBase = blockIdx.y * 128;
    const int dBase   = blockIdx.x * 64;
    const float* Wde = Wd + (size_t)e * DFF_ * D_;

    v8f acc[2][2];
    const v8f vzero = {0.f, 0.f, 0.f, 0.f, 0.f, 0.f, 0.f, 0.f};
#pragma unroll
    for (int i = 0; i < 2; ++i)
#pragma unroll
        for (int j = 0; j < 2; ++j) acc[i][j] = vzero;

    const int bk  = tid >> 3;              // 0..31
    const int bn0 = (tid & 7) * 8;
#if !USE_TDM
    const int arow0 = tid >> 2;            // 0..63 (x2 iterations)
    const int acol  = (tid & 3) * 8;       // bf16 chunk of 8
#endif

    for (int kBase = 0; kBase < DFF_; kBase += 32) {
        __syncthreads();                   // prior-iteration LDS reads done
#if USE_TDM
        // ---- wave 0 issues a TDM 2D tile load: h[tokBase:+128, kBase:+32]
        if (tid < 32) {
            const unsigned ldsA = (unsigned)(uintptr_t)&Ah[0][0]; // low32 = LDS addr
            const unsigned long long ga =
                (unsigned long long)(uintptr_t)(h + (size_t)tokBase * DFF_ + kBase);
            u32x4 g0;
            g0.x = 1u;                                        // count=1 valid D#
            g0.y = ldsA;                                      // lds_addr (bytes)
            g0.z = (unsigned)(ga & 0xFFFFFFFFu);              // global_addr[31:0]
            g0.w = (unsigned)((ga >> 32) & 0x1FFFFFFu)        // global_addr[56:32]
                 | (2u << 30);                                // type = 2 (image)
            i32x8 g1;
            g1[0] = (1 << 16)        // data_size = 2 bytes (bf16)
                  | (1 << 20)        // pad_enable: pad LDS rows
                  | (3 << 22)        // pad_interval = 16 DWORDs (64B tile row)
                  | (3 << 25);       // pad_amount  =  4 DWORDs (64B -> 80B row)
            g1[1] = (int)(((unsigned)DFF_ & 0xFFFFu) << 16);  // tensor_dim0 lo16
            g1[2] = (int)(((unsigned)DFF_ >> 16) |
                          (((unsigned)T_ & 0xFFFFu) << 16));  // dim0 hi | dim1 lo
            g1[3] = (int)(((unsigned)T_ >> 16) | (32u << 16));// dim1 hi | tile_dim0=32
            g1[4] = 128;                                      // tile_dim1=128, tile_dim2=0
            g1[5] = DFF_;                                     // tensor_dim0_stride lo32
            g1[6] = 0;                                        // stride hi | dim1_stride
            g1[7] = 0;
            const i32x4 z4 = {0, 0, 0, 0};
#if __clang_major__ >= 23
            const i32x8 z8 = {0, 0, 0, 0, 0, 0, 0, 0};
            __builtin_amdgcn_tensor_load_to_lds(g0, g1, z4, z4, z8, 0);
#else
            __builtin_amdgcn_tensor_load_to_lds(g0, g1, z4, z4, 0);
#endif
        }
#else
        // ---- stage h tile (already bf16): 128 x 32, manual path
#pragma unroll
        for (int it = 0; it < 2; ++it) {
            const int r = arow0 + it * 64;
            const uint4 hv =
                *(const uint4*)&h[(size_t)(tokBase + r) * DFF_ + kBase + acol];
            *(uint4*)&Ah[r][acol] = hv;
        }
#endif
        // ---- stage Wd tile (f32 -> bf16, transposed): 32 x 64 (overlaps TDM)
        {
            const float* ws = &Wde[(size_t)(kBase + bk) * D_ + dBase + bn0];
            const float4 w0 = *(const float4*)(ws + 0);
            const float4 w1 = *(const float4*)(ws + 4);
            Bt[bn0 + 0][bk] = (__bf16)w0.x; Bt[bn0 + 1][bk] = (__bf16)w0.y;
            Bt[bn0 + 2][bk] = (__bf16)w0.z; Bt[bn0 + 3][bk] = (__bf16)w0.w;
            Bt[bn0 + 4][bk] = (__bf16)w1.x; Bt[bn0 + 5][bk] = (__bf16)w1.y;
            Bt[bn0 + 6][bk] = (__bf16)w1.z; Bt[bn0 + 7][bk] = (__bf16)w1.w;
            if (kBase + 32 < DFF_) __builtin_prefetch(ws + 32 * D_, 0, 1);
        }
#if USE_TDM
        if (tid < 32) {                    // issuing wave drains the DMA
#if __has_builtin(__builtin_amdgcn_s_wait_tensorcnt)
            __builtin_amdgcn_s_wait_tensorcnt(0);
#else
            asm volatile("s_wait_tensorcnt 0x0" ::: "memory");
#endif
        }
#endif
        __syncthreads();                   // Ah (TDM) + Bt (ds) visible to all

        const int rsel = lane & 15;
        const int kb   = (lane >> 4) * 8;
        const int kbb  = (lane >> 4) * 8;
        BFrag a[2], b[2];
#pragma unroll
        for (int i = 0; i < 2; ++i) {
            const int r = wm * 32 + i * 16 + rsel;
            a[i].q[0] = *(const uint4*)&Ah[r][kb];
            a[i].q[1] = *(const uint4*)&Ah[r][16 + kb];
        }
#pragma unroll
        for (int j = 0; j < 2; ++j) {
            const int n = wn * 32 + j * 16 + rsel;
            b[j].q[0] = *(const uint4*)&Bt[n][kbb];
            b[j].q[1] = *(const uint4*)&Bt[n][kbb + 16];
        }
#pragma unroll
        for (int i = 0; i < 2; ++i)
#pragma unroll
            for (int j = 0; j < 2; ++j)
                acc[i][j] = __builtin_amdgcn_wmma_f32_16x16x32_bf16(
                    false, a[i].v, false, b[j].v, (short)0, acc[i][j], false, false);
    }

    const int rsel  = lane & 15;
    const int rhalf = (lane >> 4) * 8;
#pragma unroll
    for (int i = 0; i < 2; ++i)
#pragma unroll
        for (int j = 0; j < 2; ++j)
#pragma unroll
            for (int v = 0; v < 8; ++v) {
                const int row = tokBase + wm * 32 + i * 16 + rhalf + v;
                const int col = dBase + wn * 32 + j * 16 + rsel;
                const size_t idx = (size_t)row * D_ + col;
                const float val = acc[i][j][v];
                if (first) out[idx] = val;       // expert 0 overwrites poison
                else       out[idx] += val;      // launches are stream-ordered
            }
}

// ---------------------------------------------------------------- launch ----
extern "C" void kernel_launch(void* const* d_in, const int* in_sizes, int n_in,
                              void* d_out, int out_size, void* d_ws, size_t ws_size,
                              hipStream_t stream) {
    (void)in_sizes; (void)n_in; (void)out_size; (void)ws_size;
    const float* x  = (const float*)d_in[0];
    const float* Wr = (const float*)d_in[1];
    const float* Wg = (const float*)d_in[2];
    const float* Wu = (const float*)d_in[3];
    const float* Wd = (const float*)d_in[4];
    float* out = (float*)d_out;

    float*  combine = (float*)d_ws;                                       // T*E f32
    __bf16* h = (__bf16*)((char*)d_ws + (size_t)T_ * E_ * sizeof(float)); // T*DFF bf16

    moe_router<<<T_ / 8, 256, 0, stream>>>(x, Wr, combine);

    const dim3 gG(DFF_ / 64, T_ / 128);   // 32 x 64
    const dim3 gD(D_ / 64,   T_ / 128);   // 16 x 64
    for (int e = 0; e < E_; ++e) {
        moe_gate_up<<<gG, 256, 0, stream>>>(x, Wg, Wu, combine, h, e);
        moe_down<<<gD, 256, 0, stream>>>(h, Wd, out, e, e == 0 ? 1 : 0);
    }
}
